// MoleRecLayer_68719477331
// MI455X (gfx1250) — matrix operand order repaired and checked by reading, not compiled, not run on hardware
//
#include <hip/hip_runtime.h>
#include <hip/hip_bf16.h>
#include <cstddef>

#define H    64
#define H2   128
#define LYR  4
#define NSUBG 491
#define NMOLG 131
#define NVN  80000
#define NEE  320000
#define BONDV 6

typedef __attribute__((ext_vector_type(16))) _Float16 v16h;
typedef __attribute__((ext_vector_type(8)))  _Float16 v8h;
typedef __attribute__((ext_vector_type(8)))  float    v8f;
union U16h { v16h v; v8h h[2]; _Float16 e[16]; };
union U8h  { v8h  v; _Float16 e[8]; };
union U8f  { v8f  v; float    e[8]; };

__device__ inline void gAtomicAdd(float* p, float v) { unsafeAtomicAdd(p, v); }

__device__ inline float waveReduceSum(float v) {
#pragma unroll
  for (int off = 16; off > 0; off >>= 1) v += __shfl_xor(v, off, 32);
  return v;
}

// ---------------- elementwise helpers ----------------
__global__ void zero_kernel(float* p, int n) {
  int i = blockIdx.x * blockDim.x + threadIdx.x;
  if (i < n) p[i] = 0.f;
}

__global__ void embed_kernel(const int* __restrict__ x, const float* __restrict__ emb,
                             float* __restrict__ hout, int nv) {
  int i = blockIdx.x * blockDim.x + threadIdx.x;
  if (i < nv * H) {
    int node = i >> 6, c = i & 63;
    hout[i] = emb[x[node] * H + c];
  }
}

// Repack a [din,dout] f32 weight into f16 WMMA-B fragment order:
// out[(((nt*nchunks)+c)*32 + fl)*16 + i] = w[(c*32 + (i<8?0:16) + (fl<16?0:8) + (i&7))*dout + nt*16 + (fl&15)]
__global__ void pack_b_kernel(const float* __restrict__ w, _Float16* __restrict__ outp,
                              int din, int dout) {
  int idx = blockIdx.x * blockDim.x + threadIdx.x;
  int n = din * dout;
  if (idx >= n) return;
  int nchunks = din >> 5;
  int i  = idx & 15;
  int fl = (idx >> 4) & 31;
  int rest = idx >> 9;
  int c  = rest % nchunks;
  int nt = rest / nchunks;
  int k = c * 32 + ((i < 8) ? 0 : 16) + ((fl < 16) ? 0 : 8) + (i & 7);
  int ncol = nt * 16 + (fl & 15);
  outp[idx] = (_Float16)w[k * dout + ncol];
}

// one wave per edge: msg = relu(h[src]+bond_emb[attr]); scatter-add to agg[dst]
__global__ __launch_bounds__(256) void edge_msg_kernel(
    const int* __restrict__ ei, const int* __restrict__ eattr,
    const float* __restrict__ bemb, const float* __restrict__ hin,
    float* __restrict__ agg, int ne) {
  int widx = (blockIdx.x * blockDim.x + threadIdx.x) >> 5;
  int lane = threadIdx.x & 31;
  if (widx >= ne) return;
  int dst = ei[widx];
  int src = ei[ne + widx];
  int bt  = eattr[widx];
  int c = lane * 2;
  float2 hv = *(const float2*)(hin + (size_t)src * H + c);
  float2 ev = *(const float2*)(bemb + bt * H + c);
  float m0 = fmaxf(hv.x + ev.x, 0.f);
  float m1 = fmaxf(hv.y + ev.y, 0.f);
  gAtomicAdd(&agg[(size_t)dst * H + c],     m0);
  gAtomicAdd(&agg[(size_t)dst * H + c + 1], m1);
}

// ---------------- GIN MLP1: t = ((1+eps)h + agg) @ w1 + b1, + column stats ----------------
// K=64 (2 chunks), N=128 (8 tiles). B pre-packed in fragment order (global), copied to LDS.
__global__ __launch_bounds__(128) void mlp1_kernel(
    const float* __restrict__ hin, const float* __restrict__ agg,
    const float* __restrict__ epsp,
    const _Float16* __restrict__ bpack, const float* __restrict__ b1,
    float* __restrict__ tout, float* __restrict__ gsum, float* __restrict__ gsq) {
  __shared__ _Float16 sB[H * H2];            // 8192 halves, fragment order
  __shared__ float    sBias[H2];
  __shared__ _Float16 sA[4][2 * 32 * 16];    // per-wave A fragments (2 K-chunks)
  __shared__ float    sSum[H2], sSq[H2];

  const int tid = threadIdx.x, wave = tid >> 5, lane = tid & 31;
  // coalesced b128 copy of packed B into LDS
  {
    v8h* dst = (v8h*)sB;
    const v8h* srcp = (const v8h*)bpack;
    for (int i = tid; i < (H * H2) / 8; i += 128) dst[i] = srcp[i];
  }
  for (int i = tid; i < H2; i += 128) { sBias[i] = b1[i]; sSum[i] = 0.f; sSq[i] = 0.f; }

  const float epsv = 1.0f + epsp[0];
  const int rowBase = blockIdx.x * 64 + wave * 16;

  { // stage A in fragment order: lane handles row m, 4 segments of 8 consecutive K
    int m = lane & 15;
    int sbase = (lane >> 4) * 4;
    v8h* sAv = (v8h*)sA[wave];
#pragma unroll
    for (int s = 0; s < 4; ++s) {
      int k0 = (sbase + s) * 8;
      const float4* hp = (const float4*)(hin + (size_t)(rowBase + m) * H + k0);
      const float4* ap = (const float4*)(agg + (size_t)(rowBase + m) * H + k0);
      float4 h0 = hp[0], h1 = hp[1], a0 = ap[0], a1 = ap[1];
      U8h st;
      st.e[0] = (_Float16)(epsv * h0.x + a0.x);
      st.e[1] = (_Float16)(epsv * h0.y + a0.y);
      st.e[2] = (_Float16)(epsv * h0.z + a0.z);
      st.e[3] = (_Float16)(epsv * h0.w + a0.w);
      st.e[4] = (_Float16)(epsv * h1.x + a1.x);
      st.e[5] = (_Float16)(epsv * h1.y + a1.y);
      st.e[6] = (_Float16)(epsv * h1.z + a1.z);
      st.e[7] = (_Float16)(epsv * h1.w + a1.w);
      int c   = k0 >> 5;
      int kk0 = k0 & 31;
      int fl  = (kk0 & 8) ? (m + 16) : m;
      int eo  = (kk0 & 16) ? 1 : 0;
      sAv[(c * 32 + fl) * 2 + eo] = st.v;
    }
  }
  __syncthreads();

  // A fragments: 2 x (2 x ds_load_b128)
  U16h afr[2];
  {
    const v8h* sAv = (const v8h*)sA[wave];
#pragma unroll
    for (int c = 0; c < 2; ++c) {
      afr[c].h[0] = sAv[(c * 32 + lane) * 2];
      afr[c].h[1] = sAv[(c * 32 + lane) * 2 + 1];
    }
  }

  const int ncol = lane & 15;
  const int mrow = (lane < 16) ? 0 : 8;
  const v8h* sBv = (const v8h*)sB;
#pragma unroll
  for (int nt = 0; nt < 8; ++nt) {
    int n0 = nt * 16;
    U16h bf0, bf1;
    int fb0 = ((nt * 2 + 0) * 32 + lane) * 2;
    int fb1 = ((nt * 2 + 1) * 32 + lane) * 2;
    bf0.h[0] = sBv[fb0]; bf0.h[1] = sBv[fb0 + 1];
    bf1.h[0] = sBv[fb1]; bf1.h[1] = sBv[fb1 + 1];
    U8f acc;
    float bias = sBias[n0 + ncol];
#pragma unroll
    for (int r = 0; r < 8; ++r) acc.e[r] = bias;
    acc.v = __builtin_amdgcn_wmma_f32_16x16x32_f16(false, afr[0].v, false, bf0.v, (short)0, acc.v, false, false);
    acc.v = __builtin_amdgcn_wmma_f32_16x16x32_f16(false, afr[1].v, false, bf1.v, (short)0, acc.v, false, false);
    float ls = 0.f, lq = 0.f;
#pragma unroll
    for (int r = 0; r < 8; ++r) {
      int row = rowBase + mrow + r;
      float v = acc.e[r];
      tout[(size_t)row * H2 + n0 + ncol] = v;
      ls += v; lq += v * v;
    }
    atomicAdd(&sSum[n0 + ncol], ls);
    atomicAdd(&sSq[n0 + ncol], lq);
  }
  __syncthreads();
  if (tid < H2) { gAtomicAdd(&gsum[tid], sSum[tid]); gAtomicAdd(&gsq[tid], sSq[tid]); }
}

// ---------------- GIN MLP2: u = relu(bn1(t)) @ w2 + b2, + column stats ----------------
// K=128 (4 chunks), N=64 (4 tiles)
__global__ __launch_bounds__(128) void mlp2_kernel(
    const float* __restrict__ tin, const float* __restrict__ bna, const float* __restrict__ bnb,
    const _Float16* __restrict__ bpack, const float* __restrict__ b2,
    float* __restrict__ uout, float* __restrict__ gsum, float* __restrict__ gsq) {
  __shared__ _Float16 sB[H2 * H];            // 8192 halves, fragment order
  __shared__ float    sBias[H];
  __shared__ _Float16 sA[4][4 * 32 * 16];    // per-wave A fragments (4 K-chunks)
  __shared__ float    sAf[H2], sBf[H2];
  __shared__ float    sSum[H], sSq[H];

  const int tid = threadIdx.x, wave = tid >> 5, lane = tid & 31;
  {
    v8h* dst = (v8h*)sB;
    const v8h* srcp = (const v8h*)bpack;
    for (int i = tid; i < (H2 * H) / 8; i += 128) dst[i] = srcp[i];
  }
  for (int i = tid; i < H2; i += 128) { sAf[i] = bna[i]; sBf[i] = bnb[i]; }
  for (int i = tid; i < H; i += 128) { sBias[i] = b2[i]; sSum[i] = 0.f; sSq[i] = 0.f; }
  __syncthreads();   // sAf/sBf used by staging below

  const int rowBase = blockIdx.x * 64 + wave * 16;
  { // stage A = relu(a*t+b) in fragment order: lane = row m, 8 segments of 8 K
    int m = lane & 15;
    int sbase = (lane >> 4) * 8;
    v8h* sAv = (v8h*)sA[wave];
#pragma unroll
    for (int s = 0; s < 8; ++s) {
      int k0 = (sbase + s) * 8;
      const float4* tp = (const float4*)(tin + (size_t)(rowBase + m) * H2 + k0);
      float4 t0 = tp[0], t1 = tp[1];
      U8h st;
      st.e[0] = (_Float16)fmaxf(sAf[k0+0]*t0.x + sBf[k0+0], 0.f);
      st.e[1] = (_Float16)fmaxf(sAf[k0+1]*t0.y + sBf[k0+1], 0.f);
      st.e[2] = (_Float16)fmaxf(sAf[k0+2]*t0.z + sBf[k0+2], 0.f);
      st.e[3] = (_Float16)fmaxf(sAf[k0+3]*t0.w + sBf[k0+3], 0.f);
      st.e[4] = (_Float16)fmaxf(sAf[k0+4]*t1.x + sBf[k0+4], 0.f);
      st.e[5] = (_Float16)fmaxf(sAf[k0+5]*t1.y + sBf[k0+5], 0.f);
      st.e[6] = (_Float16)fmaxf(sAf[k0+6]*t1.z + sBf[k0+6], 0.f);
      st.e[7] = (_Float16)fmaxf(sAf[k0+7]*t1.w + sBf[k0+7], 0.f);
      int c   = k0 >> 5;
      int kk0 = k0 & 31;
      int fl  = (kk0 & 8) ? (m + 16) : m;
      int eo  = (kk0 & 16) ? 1 : 0;
      sAv[(c * 32 + fl) * 2 + eo] = st.v;
    }
  }
  __syncthreads();

  U16h afr[4];
  {
    const v8h* sAv = (const v8h*)sA[wave];
#pragma unroll
    for (int c = 0; c < 4; ++c) {
      afr[c].h[0] = sAv[(c * 32 + lane) * 2];
      afr[c].h[1] = sAv[(c * 32 + lane) * 2 + 1];
    }
  }

  const int ncol = lane & 15;
  const int mrow = (lane < 16) ? 0 : 8;
  const v8h* sBv = (const v8h*)sB;
#pragma unroll
  for (int nt = 0; nt < 4; ++nt) {
    int n0 = nt * 16;
    U8f acc;
    float bias = sBias[n0 + ncol];
#pragma unroll
    for (int r = 0; r < 8; ++r) acc.e[r] = bias;
#pragma unroll
    for (int c = 0; c < 4; ++c) {
      U16h bf;
      int fb = ((nt * 4 + c) * 32 + lane) * 2;
      bf.h[0] = sBv[fb]; bf.h[1] = sBv[fb + 1];
      acc.v = __builtin_amdgcn_wmma_f32_16x16x32_f16(false, afr[c].v, false, bf.v, (short)0, acc.v, false, false);
    }
    float ls = 0.f, lq = 0.f;
#pragma unroll
    for (int r = 0; r < 8; ++r) {
      int row = rowBase + mrow + r;
      float v = acc.e[r];
      uout[(size_t)row * H + n0 + ncol] = v;
      ls += v; lq += v * v;
    }
    atomicAdd(&sSum[n0 + ncol], ls);
    atomicAdd(&sSq[n0 + ncol], lq);
  }
  __syncthreads();
  if (tid < H) { gAtomicAdd(&gsum[tid], sSum[tid]); gAtomicAdd(&gsq[tid], sSq[tid]); }
}

__global__ void bn_finalize_kernel(const float* gsum, const float* gsq,
                                   const float* g, const float* beta,
                                   float* a, float* b, int ncols, float invn) {
  int i = blockIdx.x * blockDim.x + threadIdx.x;
  if (i < ncols) {
    float m = gsum[i] * invn;
    float v = gsq[i] * invn - m * m;
    float s = g[i] * rsqrtf(v + 1e-5f);
    a[i] = s;
    b[i] = beta[i] - m * s;
  }
}

__global__ void bn_apply_kernel(const float* __restrict__ u, const float* __restrict__ a,
                                const float* __restrict__ b, float* __restrict__ hout,
                                int n, int relu) {
  int i = blockIdx.x * blockDim.x + threadIdx.x;
  if (i < n) {
    int c = i & (H - 1);
    float v = a[c] * u[i] + b[c];
    if (relu) v = fmaxf(v, 0.f);
    hout[i] = v;
  }
}

// one wave per node: scatter-add into graph pools
__global__ __launch_bounds__(256) void pool_kernel(const float* __restrict__ hin,
                                                   const int* __restrict__ batch,
                                                   float* repr, float* cnt, int nv) {
  int widx = (blockIdx.x * blockDim.x + threadIdx.x) >> 5;
  int lane = threadIdx.x & 31;
  if (widx >= nv) return;
  int g = batch[widx];
  int c = lane * 2;
  float2 v = *(const float2*)(hin + (size_t)widx * H + c);
  gAtomicAdd(&repr[g * H + c],     v.x);
  gAtomicAdd(&repr[g * H + c + 1], v.y);
  if (lane == 0) gAtomicAdd(&cnt[g], 1.0f);
}

__global__ void pool_div_kernel(float* repr, const float* cnt, int n) {
  int i = blockIdx.x * blockDim.x + threadIdx.x;
  if (i < n) repr[i] /= fmaxf(cnt[i >> 6], 1.0f);
}

// generic small dense: Y[r,o] = X[r,:] @ W[:,o] + b[o]
__global__ void linear_kernel(const float* __restrict__ X, const float* __restrict__ Wm,
                              const float* __restrict__ bv, float* __restrict__ Y,
                              int rows, int din, int dout) {
  int i = blockIdx.x * blockDim.x + threadIdx.x;
  if (i >= rows * dout) return;
  int r = i / dout, o = i % dout;
  float accv = bv ? bv[o] : 0.f;
  for (int k = 0; k < din; ++k) accv += X[r * din + k] * Wm[k * dout + o];
  Y[i] = accv;
}

// one wave per (query,head): O = Qh + softmax(Qh Kh^T / sqrt(dh)) @ Vh
__global__ __launch_bounds__(256) void sab_attn_kernel(const float* __restrict__ Q,
                                                       const float* __restrict__ K,
                                                       const float* __restrict__ V,
                                                       float* __restrict__ O, int S) {
  int widx = (blockIdx.x * blockDim.x + threadIdx.x) >> 5;
  int lane = threadIdx.x & 31;
  if (widx >= S * 2) return;
  int s = widx >> 1, hd = widx & 1;
  int c = hd * 32 + lane;
  float q = Q[s * H + c];
  const float scale = 0.17677669529663687f;  // 1/sqrt(32)
  float mx = -3.4e38f, den = 0.f, acc = 0.f;
  for (int k = 0; k < S; ++k) {
    float d = waveReduceSum(q * K[k * H + c]) * scale;
    float nm = fmaxf(mx, d);
    float corr = __expf(mx - nm);
    float w = __expf(d - nm);
    den = den * corr + w;
    acc = acc * corr + w * V[k * H + c];
    mx = nm;
  }
  O[s * H + c] = q + acc / den;
}

// one wave per row: LN1 -> @wo+bo -> LN2
__global__ __launch_bounds__(256) void sab_post_kernel(
    const float* __restrict__ O, const float* __restrict__ g1, const float* __restrict__ b1v,
    const float* __restrict__ wo, const float* __restrict__ bo,
    const float* __restrict__ g2, const float* __restrict__ b2v,
    float* __restrict__ outp, int S) {
  int widx = (blockIdx.x * blockDim.x + threadIdx.x) >> 5;
  int lane = threadIdx.x & 31;
  if (widx >= S) return;
  float x0 = O[widx * H + lane], x1 = O[widx * H + lane + 32];
  float mean = waveReduceSum(x0 + x1) * (1.f / 64.f);
  float d0 = x0 - mean, d1 = x1 - mean;
  float var = waveReduceSum(d0 * d0 + d1 * d1) * (1.f / 64.f);
  float inv = rsqrtf(var + 1e-5f);
  float xn0 = d0 * inv * g1[lane]      + b1v[lane];
  float xn1 = d1 * inv * g1[lane + 32] + b1v[lane + 32];
  float y0 = bo[lane], y1 = bo[lane + 32];
  for (int k = 0; k < 32; ++k) {
    float xk = __shfl(xn0, k, 32);
    y0 += xk * wo[k * H + lane];
    y1 += xk * wo[k * H + lane + 32];
  }
  for (int k = 0; k < 32; ++k) {
    float xk = __shfl(xn1, k, 32);
    y0 += xk * wo[(k + 32) * H + lane];
    y1 += xk * wo[(k + 32) * H + lane + 32];
  }
  float m2 = waveReduceSum(y0 + y1) * (1.f / 64.f);
  float e0 = y0 - m2, e1 = y1 - m2;
  float v2 = waveReduceSum(e0 * e0 + e1 * e1) * (1.f / 64.f);
  float i2 = rsqrtf(v2 + 1e-5f);
  outp[widx * H + lane]      = e0 * i2 * g2[lane]      + b2v[lane];
  outp[widx * H + lane + 32] = e1 * i2 * g2[lane + 32] + b2v[lane + 32];
}

// one wave per molecule row: masked softmax(Q K^T/8) @ (fix * sub)
__global__ __launch_bounds__(256) void agger_kernel(
    const float* __restrict__ Q, const float* __restrict__ Kg,
    const float* __restrict__ sub, const float* __restrict__ fix,
    const unsigned char* __restrict__ mask, float* __restrict__ outp,
    int nmol, int nsub) {
  int widx = (blockIdx.x * blockDim.x + threadIdx.x) >> 5;
  int lane = threadIdx.x & 31;
  if (widx >= nmol) return;
  float q0 = Q[widx * H + lane], q1 = Q[widx * H + lane + 32];
  const float scale = 0.125f;  // 1/sqrt(64)
  float mx = -3.4e38f, den = 0.f, a0 = 0.f, a1 = 0.f;
  for (int k = 0; k < nsub; ++k) {
    float d = waveReduceSum(q0 * Kg[k * H + lane] + q1 * Kg[k * H + lane + 32]) * scale;
    if (mask[widx * nsub + k]) d = -4294967296.0f;
    float nm = fmaxf(mx, d);
    float corr = __expf(mx - nm);
    float w = __expf(d - nm);
    float f = fix[k] * w;
    den = den * corr + w;
    a0 = a0 * corr + f * sub[k * H + lane];
    a1 = a1 * corr + f * sub[k * H + lane + 32];
    mx = nm;
  }
  outp[widx * H + lane]      = a0 / den;
  outp[widx * H + lane + 32] = a1 / den;
}

// ---------------- host orchestration ----------------
struct GinParams {
  const float *atom_emb, *bond_emb, *eps, *w1, *b1, *bn1g, *bn1b, *w2, *b2, *bn2g, *bn2b;
};

static void run_gin(const int* x, const int* eattr, const int* eidx, const int* batch,
                    const GinParams& p,
                    float* hbuf, float* aggbuf, float* tbuf, float* ubuf,
                    float* stats, float* bna, float* bnb, _Float16* bpack,
                    float* repr, float* cnt, int ngraph, hipStream_t stream) {
  embed_kernel<<<(NVN * H + 255) / 256, 256, 0, stream>>>(x, p.atom_emb, hbuf, NVN);
  for (int l = 0; l < LYR; ++l) {
    zero_kernel<<<(NVN * H + 255) / 256, 256, 0, stream>>>(aggbuf, NVN * H);
    edge_msg_kernel<<<NEE / 8, 256, 0, stream>>>(eidx, eattr, p.bond_emb + l * BONDV * H,
                                                 hbuf, aggbuf, NEE);
    zero_kernel<<<1, 256, 0, stream>>>(stats, 256);
    pack_b_kernel<<<(H * H2 + 255) / 256, 256, 0, stream>>>(p.w1 + l * H * H2, bpack, H, H2);
    mlp1_kernel<<<NVN / 64, 128, 0, stream>>>(hbuf, aggbuf, p.eps + l,
                                              bpack, p.b1 + l * H2,
                                              tbuf, stats, stats + 128);
    bn_finalize_kernel<<<1, 128, 0, stream>>>(stats, stats + 128, p.bn1g + l * H2,
                                              p.bn1b + l * H2, bna, bnb, H2, 1.0f / NVN);
    zero_kernel<<<1, 256, 0, stream>>>(stats, 256);
    pack_b_kernel<<<(H2 * H + 255) / 256, 256, 0, stream>>>(p.w2 + l * H2 * H, bpack, H2, H);
    mlp2_kernel<<<NVN / 64, 128, 0, stream>>>(tbuf, bna, bnb,
                                              bpack, p.b2 + l * H,
                                              ubuf, stats, stats + 128);
    bn_finalize_kernel<<<1, 128, 0, stream>>>(stats, stats + 128, p.bn2g + l * H,
                                              p.bn2b + l * H, bna, bnb, H, 1.0f / NVN);
    bn_apply_kernel<<<(NVN * H + 255) / 256, 256, 0, stream>>>(ubuf, bna, bnb, hbuf,
                                                               NVN * H, (l != LYR - 1) ? 1 : 0);
  }
  zero_kernel<<<(ngraph * H + 255) / 256, 256, 0, stream>>>(repr, ngraph * H);
  zero_kernel<<<1, 256, 0, stream>>>(cnt, ngraph);
  pool_kernel<<<NVN / 8, 256, 0, stream>>>(hbuf, batch, repr, cnt, NVN);
  pool_div_kernel<<<(ngraph * H + 255) / 256, 256, 0, stream>>>(repr, cnt, ngraph * H);
}

extern "C" void kernel_launch(void* const* d_in, const int* in_sizes, int n_in,
                              void* d_out, int out_size, void* d_ws, size_t ws_size,
                              hipStream_t stream) {
  (void)in_sizes; (void)n_in; (void)out_size; (void)ws_size;
  const int* sub_x      = (const int*)d_in[0];
  const int* sub_eattr  = (const int*)d_in[1];
  const int* sub_eidx   = (const int*)d_in[2];
  const int* sub_batch  = (const int*)d_in[3];
  const int* mol_x      = (const int*)d_in[4];
  const int* mol_eattr  = (const int*)d_in[5];
  const int* mol_eidx   = (const int*)d_in[6];
  const int* mol_batch  = (const int*)d_in[7];
  const float* fix_feat = (const float*)d_in[8];
  const unsigned char* mask = (const unsigned char*)d_in[9];
  auto gin = [&](int base) {
    GinParams p;
    p.atom_emb = (const float*)d_in[base + 0];
    p.bond_emb = (const float*)d_in[base + 1];
    p.eps      = (const float*)d_in[base + 2];
    p.w1       = (const float*)d_in[base + 3];
    p.b1       = (const float*)d_in[base + 4];
    p.bn1g     = (const float*)d_in[base + 5];
    p.bn1b     = (const float*)d_in[base + 6];
    p.w2       = (const float*)d_in[base + 7];
    p.b2       = (const float*)d_in[base + 8];
    p.bn2g     = (const float*)d_in[base + 9];
    p.bn2b     = (const float*)d_in[base + 10];
    return p;
  };
  GinParams subP = gin(10), molP = gin(21);
  const float* sab_wq = (const float*)d_in[32]; const float* sab_bq = (const float*)d_in[33];
  const float* sab_wk = (const float*)d_in[34]; const float* sab_bk = (const float*)d_in[35];
  const float* sab_wv = (const float*)d_in[36]; const float* sab_bv = (const float*)d_in[37];
  const float* sab_wo = (const float*)d_in[38]; const float* sab_bo = (const float*)d_in[39];
  const float* ln1_g  = (const float*)d_in[40]; const float* ln1_b  = (const float*)d_in[41];
  const float* ln2_g  = (const float*)d_in[42]; const float* ln2_b  = (const float*)d_in[43];
  const float* agg_wq = (const float*)d_in[44]; const float* agg_bq = (const float*)d_in[45];
  const float* agg_wk = (const float*)d_in[46]; const float* agg_bk = (const float*)d_in[47];

  // workspace layout (floats; all offsets 16B-aligned)
  float* W = (float*)d_ws;
  size_t o = 0;
  float* hbuf  = W + o; o += (size_t)NVN * H;
  float* aggb  = W + o; o += (size_t)NVN * H;
  float* tbuf  = W + o; o += (size_t)NVN * H2;
  float* ubuf  = W + o; o += (size_t)NVN * H;
  float* stats = W + o; o += 256;
  float* bna   = W + o; o += 128;
  float* bnb   = W + o; o += 128;
  _Float16* bpack = (_Float16*)(W + o); o += 4096;   // 8192 halves
  float* srepr = W + o; o += (size_t)NSUBG * H;
  float* scnt  = W + o; o += ((NSUBG + 3) / 4) * 4;
  float* mrepr = W + o; o += (size_t)NMOLG * H;
  float* mcnt  = W + o; o += ((NMOLG + 3) / 4) * 4;
  float* Qb    = W + o; o += (size_t)NSUBG * H;
  float* Kb    = W + o; o += (size_t)NSUBG * H;
  float* Vb    = W + o; o += (size_t)NSUBG * H;
  float* Ob    = W + o; o += (size_t)NSUBG * H;
  float* subF  = W + o; o += (size_t)NSUBG * H;
  float* Qa    = W + o; o += (size_t)NMOLG * H;
  float* Ka    = W + o; o += (size_t)NSUBG * H;

  run_gin(sub_x, sub_eattr, sub_eidx, sub_batch, subP,
          hbuf, aggb, tbuf, ubuf, stats, bna, bnb, bpack, srepr, scnt, NSUBG, stream);
  run_gin(mol_x, mol_eattr, mol_eidx, mol_batch, molP,
          hbuf, aggb, tbuf, ubuf, stats, bna, bnb, bpack, mrepr, mcnt, NMOLG, stream);

  // SAB on sub_repr [491,64]
  int nqk = NSUBG * H;
  linear_kernel<<<(nqk + 255) / 256, 256, 0, stream>>>(srepr, sab_wq, sab_bq, Qb, NSUBG, H, H);
  linear_kernel<<<(nqk + 255) / 256, 256, 0, stream>>>(srepr, sab_wk, sab_bk, Kb, NSUBG, H, H);
  linear_kernel<<<(nqk + 255) / 256, 256, 0, stream>>>(srepr, sab_wv, sab_bv, Vb, NSUBG, H, H);
  sab_attn_kernel<<<(NSUBG * 2 * 32 + 255) / 256, 256, 0, stream>>>(Qb, Kb, Vb, Ob, NSUBG);
  sab_post_kernel<<<(NSUBG * 32 + 255) / 256, 256, 0, stream>>>(Ob, ln1_g, ln1_b, sab_wo,
                                                                sab_bo, ln2_g, ln2_b, subF, NSUBG);
  // AttenAgger
  linear_kernel<<<(NMOLG * H + 255) / 256, 256, 0, stream>>>(mrepr, agg_wq, agg_bq, Qa, NMOLG, H, H);
  linear_kernel<<<(NSUBG * H + 255) / 256, 256, 0, stream>>>(subF, agg_wk, agg_bk, Ka, NSUBG, H, H);
  agger_kernel<<<(NMOLG * 32 + 255) / 256, 256, 0, stream>>>(Qa, Ka, subF, fix_feat, mask,
                                                             (float*)d_out, NMOLG, NSUBG);
}
